// StudentQAT_50852412784643
// MI455X (gfx1250) — 8-bit matrix operand order repaired and checked by reading, compiled, not run on hardware
//
#include <hip/hip_runtime.h>

typedef __attribute__((ext_vector_type(8))) int v8i;

#define BATCH 8192

__device__ __forceinline__ unsigned sgnbyte_f(float s) { return (s >= 0.0f) ? 0x01u : 0xFFu; }
__device__ __forceinline__ signed char sgnbyte_i(int s) { return (s >= 0) ? (signed char)1 : (signed char)-1; }

__global__ void k_quant_w1(const float* __restrict__ w1, float* __restrict__ wq) {
    __shared__ float red[256];
    int t = threadIdx.x;
    float m = 0.0f;
    if (t < 144) m = fabsf(w1[t]);
    red[t] = m;
    __syncthreads();
    for (int s = 128; s > 0; s >>= 1) {
        if (t < s) red[t] = fmaxf(red[t], red[t + s]);
        __syncthreads();
    }
    float scale = red[0] / 127.0f + 1e-8f;
    if (t < 144) {
        float q = rintf(w1[t] / scale);
        q = fminf(fmaxf(q, -128.0f), 127.0f);
        wq[t] = q;
    }
}

__global__ void k_conv1(const float* __restrict__ x, const float* __restrict__ wq,
                        signed char* __restrict__ s1) {
    __shared__ float w[144];
    if (threadIdx.x < 144) w[threadIdx.x] = wq[threadIdx.x];
    __syncthreads();
    long tid = (long)blockIdx.x * 256 + threadIdx.x;
    if (tid >= (long)BATCH * 26 * 26) return;
    int b = (int)(tid / 676);
    int rem = (int)(tid % 676);
    int y = rem / 26, xx = rem % 26;
    const float* xb = x + ((long)b * 28 + y) * 28 + xx;
    float in[9];
#pragma unroll
    for (int ky = 0; ky < 3; ky++)
#pragma unroll
        for (int kx = 0; kx < 3; kx++) in[ky * 3 + kx] = xb[ky * 28 + kx];
    int outd[4];
#pragma unroll
    for (int cg = 0; cg < 4; cg++) {
        unsigned r = 0;
#pragma unroll
        for (int j = 0; j < 4; j++) {
            int c = cg * 4 + j;
            float s = 0.0f;
#pragma unroll
            for (int k = 0; k < 9; k++) s += in[k] * w[c * 9 + k];
            r |= sgnbyte_f(s) << (8 * j);
        }
        outd[cg] = (int)r;
    }
    *(int4*)(s1 + tid * 16) = make_int4(outd[0], outd[1], outd[2], outd[3]);
}


__global__ void k_pack_w2(const float* __restrict__ w2, int* __restrict__ p2) {
    int t = blockIdx.x * 256 + threadIdx.x;
    if (t >= 768) return;
    int ck = t >> 8, v = (t >> 5) & 7, lane = t & 31;
    int h = lane >> 4, n = lane & 15;
    unsigned r = 0;
#pragma unroll
    for (int j = 0; j < 4; j++) {
        int K = ck * 64 + (v >> 1) * 16 + h * 8 + (v & 1) * 4 + j;
        unsigned byte = 0;
        if (K < 144) {
            int tap = K >> 4, c = K & 15, ky = tap / 3, kx = tap % 3;
            byte = sgnbyte_f(w2[((n * 16 + c) * 3 + ky) * 3 + kx]);
        }
        r |= byte << (8 * j);
    }
    p2[t] = (int)r;
}

__global__ void k_pack_w3(const float* __restrict__ w3, int* __restrict__ p3) {
    int t = blockIdx.x * 256 + threadIdx.x;
    if (t >= 1536) return;
    int nh = t / 768, u = t % 768;
    int ck = u >> 8, v = (u >> 5) & 7, lane = u & 31;
    int h = lane >> 4, n = nh * 16 + (lane & 15);
    unsigned r = 0;
#pragma unroll
    for (int j = 0; j < 4; j++) {
        int K = ck * 64 + (v >> 1) * 16 + h * 8 + (v & 1) * 4 + j;
        unsigned byte = 0;
        if (K < 144) {
            int tap = K >> 4, c = K & 15, ky = tap / 3, kx = tap % 3;
            byte = sgnbyte_f(w3[((n * 16 + c) * 3 + ky) * 3 + kx]);
        }
        r |= byte << (8 * j);
    }
    p3[t] = (int)r;
}

__global__ void k_pack_wfc(const float* __restrict__ wfc, int* __restrict__ pfc) {
    int t = blockIdx.x * 256 + threadIdx.x;
    if (t >= 61 * 256) return;
    int ck = t >> 8, v = (t >> 5) & 7, lane = t & 31;
    int h = lane >> 4, n = lane & 15;
    unsigned r = 0;
#pragma unroll
    for (int j = 0; j < 4; j++) {
        int K = ck * 64 + (v >> 1) * 16 + h * 8 + (v & 1) * 4 + j;
        unsigned byte = 0;
        if (K < 3872 && n < 10) {
            int pos = K >> 5, c = K & 31;
            byte = sgnbyte_f(wfc[n * 3872 + c * 121 + pos]);
        }
        r |= byte << (8 * j);
    }
    pfc[t] = (int)r;
}


__global__ void k_conv2(const signed char* __restrict__ s1, const int* __restrict__ p2,
                        signed char* __restrict__ s2) {
    __shared__ signed char smem[8 * 256];
    int lane = threadIdx.x & 31;
    int wv = threadIdx.x >> 5;
    int h = lane >> 4, l15 = lane & 15;
    int waveId = blockIdx.x * 8 + wv;
    int nW = gridDim.x * 8;
    v8i bf[3];
#pragma unroll
    for (int ck = 0; ck < 3; ck++)
#pragma unroll
        for (int v = 0; v < 8; v++) bf[ck][v] = p2[ck * 256 + v * 32 + lane];
    const int NT = (BATCH * 576) / 16;
    for (int tile = waveId; tile < NT; tile += nW) {
        int m = tile * 16 + l15;
        int b = m / 576, rem = m % 576;
        int y = rem / 24, xx = rem % 24;
        const signed char* base = s1 + (((long)b * 26 + y) * 26 + xx) * 16;
        v8i acc = {};
#pragma unroll
        for (int ck = 0; ck < 3; ck++) {
            v8i a;
#pragma unroll
            for (int v = 0; v < 8; v++) {
                int tap = ck * 4 + (v >> 1);
                if (tap < 9) {
                    int ky = tap / 3, kx = tap % 3;
                    a[v] = *(const int*)(base + (ky * 26 + kx) * 16 + h * 8 + (v & 1) * 4);
                } else {
                    a[v] = 0;
                }
            }
            acc = __builtin_amdgcn_wmma_i32_16x16x64_iu8(true, a, true, bf[ck], acc, false, false);
        }
        signed char* sm = smem + wv * 256;
#pragma unroll
        for (int r = 0; r < 8; r++) sm[(r + h * 8) * 16 + l15] = sgnbyte_i(acc[r]);
        asm volatile("s_wait_dscnt 0" ::: "memory");
        if (lane < 16) {
            const int* sr = (const int*)(smem + wv * 256 + lane * 16);
            *(int4*)(s2 + (long)(tile * 16 + lane) * 16) = make_int4(sr[0], sr[1], sr[2], sr[3]);
        }
    }
}

__global__ void k_conv3(const signed char* __restrict__ s2, const int* __restrict__ p3,
                        signed char* __restrict__ s3) {
    __shared__ signed char smem[8 * 512];
    int lane = threadIdx.x & 31;
    int wv = threadIdx.x >> 5;
    int h = lane >> 4, l15 = lane & 15;
    int waveId = blockIdx.x * 8 + wv;
    int nW = gridDim.x * 8;
    v8i bf[2][3];
#pragma unroll
    for (int nh = 0; nh < 2; nh++)
#pragma unroll
        for (int ck = 0; ck < 3; ck++)
#pragma unroll
            for (int v = 0; v < 8; v++) bf[nh][ck][v] = p3[(nh * 3 + ck) * 256 + v * 32 + lane];
    const int NT = (BATCH * 484) / 16;
    for (int tile = waveId; tile < NT; tile += nW) {
        int m = tile * 16 + l15;
        int b = m / 484, rem = m % 484;
        int y = rem / 22, xx = rem % 22;
        const signed char* base = s2 + (((long)b * 24 + y) * 24 + xx) * 16;
        v8i a[3];
#pragma unroll
        for (int ck = 0; ck < 3; ck++)
#pragma unroll
            for (int v = 0; v < 8; v++) {
                int tap = ck * 4 + (v >> 1);
                if (tap < 9) {
                    int ky = tap / 3, kx = tap % 3;
                    a[ck][v] = *(const int*)(base + (ky * 24 + kx) * 16 + h * 8 + (v & 1) * 4);
                } else {
                    a[ck][v] = 0;
                }
            }
        v8i acc0 = {}, acc1 = {};
#pragma unroll
        for (int ck = 0; ck < 3; ck++) {
            acc0 = __builtin_amdgcn_wmma_i32_16x16x64_iu8(true, a[ck], true, bf[0][ck], acc0, false, false);
            acc1 = __builtin_amdgcn_wmma_i32_16x16x64_iu8(true, a[ck], true, bf[1][ck], acc1, false, false);
        }
        signed char* sm = smem + wv * 512;
#pragma unroll
        for (int r = 0; r < 8; r++) {
            sm[(r + h * 8) * 32 + l15] = sgnbyte_i(acc0[r]);
            sm[(r + h * 8) * 32 + 16 + l15] = sgnbyte_i(acc1[r]);
        }
        asm volatile("s_wait_dscnt 0" ::: "memory");
        if (lane < 16) {
            const int* sr = (const int*)(smem + wv * 512 + lane * 32);
            int4* d = (int4*)(s3 + (long)(tile * 16 + lane) * 32);
            d[0] = make_int4(sr[0], sr[1], sr[2], sr[3]);
            d[1] = make_int4(sr[4], sr[5], sr[6], sr[7]);
        }
    }
}

__global__ void k_pool(const signed char* __restrict__ s3, signed char* __restrict__ pd) {
    long tid = (long)blockIdx.x * 256 + threadIdx.x;
    if (tid >= (long)BATCH * 121 * 8) return;
    int dw = (int)(tid & 7);
    long pos = tid >> 3;
    int b = (int)(pos / 121);
    int rem = (int)(pos % 121);
    int py = rem / 11, px = rem % 11;
    const signed char* base = s3 + (((long)b * 22 + py * 2) * 22 + px * 2) * 32 + dw * 4;
    int accb[4] = {0, 0, 0, 0};
#pragma unroll
    for (int dy = 0; dy < 2; dy++)
#pragma unroll
        for (int dx = 0; dx < 2; dx++) {
            int w = *(const int*)(base + (dy * 22 + dx) * 32);
#pragma unroll
            for (int j = 0; j < 4; j++) accb[j] += (signed char)(w >> (8 * j));
        }
    unsigned r = 0;
#pragma unroll
    for (int j = 0; j < 4; j++) r |= ((unsigned)(unsigned char)(signed char)accb[j]) << (8 * j);
    *(int*)(pd + tid * 4) = (int)r;
}

__global__ void k_fc(const signed char* __restrict__ pd, const int* __restrict__ pfc,
                     const float* __restrict__ bias, float* __restrict__ out) {
    int lane = threadIdx.x & 31;
    int wave = (blockIdx.x * 256 + threadIdx.x) >> 5;
    int h = lane >> 4, l15 = lane & 15;
    const signed char* base = pd + (long)(wave * 16 + l15) * 3872;
    v8i acc = {};
    for (int ck = 0; ck < 61; ck++) {
        v8i a, bfr;
#pragma unroll
        for (int v = 0; v < 8; v++) {
            int Kb = ck * 64 + (v >> 1) * 16 + h * 8 + (v & 1) * 4;
            a[v] = (Kb < 3872) ? *(const int*)(base + Kb) : 0;
            bfr[v] = pfc[ck * 256 + v * 32 + lane];
        }
        acc = __builtin_amdgcn_wmma_i32_16x16x64_iu8(true, a, true, bfr, acc, false, false);
    }
#pragma unroll
    for (int r = 0; r < 8; r++) {
        int orow = wave * 16 + r + h * 8;
        if (l15 < 10) out[orow * 10 + l15] = 0.25f * (float)acc[r] + bias[l15];
    }
}

extern "C" void kernel_launch(void* const* d_in, const int* in_sizes, int n_in,
                              void* d_out, int out_size, void* d_ws, size_t ws_size,
                              hipStream_t stream) {
    (void)in_sizes; (void)n_in; (void)out_size; (void)ws_size;
    const float* x   = (const float*)d_in[0];
    const float* w1  = (const float*)d_in[1];
    const float* w2  = (const float*)d_in[2];
    const float* w3  = (const float*)d_in[3];
    const float* wfc = (const float*)d_in[4];
    const float* bfc = (const float*)d_in[5];
    float* out = (float*)d_out;
    char* ws = (char*)d_ws;

    signed char* s13 = (signed char*)ws;
    signed char* s2  = (signed char*)(ws + 126877696);
    signed char* pd  = (signed char*)(ws + 126877696 + 75497472);
    float* wq = (float*)(ws + 126877696 + 75497472 + 31719424);
    int* p2   = (int*)((char*)wq + 1024);
    int* p3   = (int*)((char*)p2 + 3072);
    int* pfc  = (int*)((char*)p3 + 6144);

    hipLaunchKernelGGL(k_quant_w1, dim3(1), dim3(256), 0, stream, w1, wq);
    hipLaunchKernelGGL(k_pack_w2, dim3(3), dim3(256), 0, stream, w2, p2);
    hipLaunchKernelGGL(k_pack_w3, dim3(6), dim3(256), 0, stream, w3, p3);
    hipLaunchKernelGGL(k_pack_wfc, dim3(61), dim3(256), 0, stream, wfc, pfc);
    hipLaunchKernelGGL(k_conv1, dim3(21632), dim3(256), 0, stream, x, wq, s13);
    hipLaunchKernelGGL(k_conv2, dim3(4608), dim3(256), 0, stream, s13, p2, s2);
    hipLaunchKernelGGL(k_conv3, dim3(3872), dim3(256), 0, stream, s2, p3, s13);
    hipLaunchKernelGGL(k_pool, dim3(30976), dim3(256), 0, stream, s13, pd);
    hipLaunchKernelGGL(k_fc, dim3(64), dim3(256), 0, stream, pd, pfc, bfc, out);
}
